// MultiHeadAttention_24343874633944
// MI455X (gfx1250) — matrix operand order repaired and checked by reading, not compiled, not run on hardware
//
#include <hip/hip_runtime.h>

// ---------------------------------------------------------------------------
// MI455X (gfx1250) multi-head attention, bf16 WMMA pipeline.
//   1) Q/K/V projections: f32 x f32 -> bf16 head-major [B,H,S,dh] (WMMA bf16,
//      double-buffered LDS, register-staged software pipeline)
//   2) Flash attention per (b,h,q-tile): WMMA scores + online softmax + WMMA PV
//      with register-prefetched K/V tiles
//   3) Output projection: bf16 ctx x Wo + bo -> f32 token-major
// ---------------------------------------------------------------------------

typedef __attribute__((ext_vector_type(16))) __bf16 v16bf;
typedef __attribute__((ext_vector_type(8)))  float  v8f;

union FragU {
    v16bf v;
    unsigned int u[8];
};

// A-operand gather (16-bit A, 16x32 MxK layout, ISA 7.12.2):
//   lane: M = lane&15, half = lane>>4
//   element pair j : K = half*16 + 2j  (contiguous)
__device__ __forceinline__ v16bf gather_a(const __bf16* rowptr, int half) {
    FragU f;
    const int kb = half * 16;
#pragma unroll
    for (int j = 0; j < 4; ++j)
        f.u[j] = *(const unsigned int*)(rowptr + kb + 2 * j);
#pragma unroll
    for (int j = 4; j < 8; ++j)
        f.u[j] = *(const unsigned int*)(rowptr + kb + 2 * j);
    return f.v;
}

// B-operand gather (16-bit B, 32x16 KxN layout):
//   lane: N = lane&15, element e: K = (lane>>4)*16 + e  (contiguous)
//   caller pre-offsets rowptr by (kstep + half*16)
__device__ __forceinline__ v16bf gather_b(const __bf16* rowptr) {
    FragU f;
#pragma unroll
    for (int j = 0; j < 8; ++j)
        f.u[j] = *(const unsigned int*)(rowptr + 2 * j);
    return f.v;
}

__device__ __forceinline__ v8f wmma_bf16(v16bf a, v16bf b, v8f c) {
    return __builtin_amdgcn_wmma_f32_16x16x32_bf16(
        false, a, false, b, (short)0, c, false, false);
}

__device__ __forceinline__ unsigned int pack_bf16_pair(float lo, float hi) {
    unsigned int a = (unsigned int)__builtin_bit_cast(unsigned short, (__bf16)lo);
    unsigned int b = (unsigned int)__builtin_bit_cast(unsigned short, (__bf16)hi);
    return a | (b << 16);
}

// ---------------------------------------------------------------------------
// 128x128-tile GEMM:  out = (A @ W + bias) * scale
//   A: [M,K] f32 (A_F32) or bf16.  W: [K,N] f32.
//   HEAD_MAJOR: out_bf [B,H,S,dh] bf16; else out_f [M,N] f32.
// 8 waves / 256 threads. Each wave: 2 M-tiles x 4 N-tiles (32x64 region).
// Double-buffered LDS, one barrier per 32-wide k-step, register prefetch.
// ---------------------------------------------------------------------------
template <bool A_F32, bool HEAD_MAJOR>
__global__ __launch_bounds__(256) void gemm_bf16_kernel(
    const float* __restrict__ Af, const __bf16* __restrict__ Abf,
    const float* __restrict__ W, const float* __restrict__ bias,
    __bf16* __restrict__ out_bf, float* __restrict__ out_f,
    int M, int K, int N, float scale, int S)
{
    __shared__ __bf16 As[2][128][34];   // [buf][m][k]  (68B rows, 4B aligned)
    __shared__ __bf16 Bs[2][128][34];   // [buf][n][k]  (W tile transposed)

    const int tid  = threadIdx.x;
    const int lane = tid & 31;
    const int wid  = tid >> 5;
    const int half = lane >> 4;
    const int l15  = lane & 15;

    const int mWG = blockIdx.y * 128;
    const int nWG = blockIdx.x * 128;

    const int wm = wid >> 1;   // 0..3  : 32-row group
    const int wn = wid & 1;    // 0..1  : 64-col group

    // per-thread staging coordinates (k0-independent)
    int rA4[4], cA4[4];     // f32 A path: float4 chunks
    int rA2[8], cA2[8];     // bf16 A path: pair chunks
    int rW[4], nW[4];       // W path: float4 chunks, transposed store
#pragma unroll
    for (int it = 0; it < 4; ++it) {
        int idx = tid + it * 256;
        rA4[it] = idx >> 3;  cA4[it] = (idx & 7) * 4;
        rW[it]  = idx >> 5;  nW[it]  = (idx & 31) * 4;
    }
#pragma unroll
    for (int it = 0; it < 8; ++it) {
        int idx = tid + it * 256;
        rA2[it] = idx >> 4;  cA2[it] = (idx & 15) * 2;
    }

    float4       aregF[4];
    unsigned int aregB[8];
    float4       wreg[4];

    // ---- prologue: stage tile k0=0 into registers ----
    if (A_F32) {
#pragma unroll
        for (int it = 0; it < 4; ++it)
            aregF[it] = *(const float4*)(Af + (size_t)(mWG + rA4[it]) * K + cA4[it]);
    } else {
#pragma unroll
        for (int it = 0; it < 8; ++it)
            aregB[it] = *(const unsigned int*)(Abf + (size_t)(mWG + rA2[it]) * K + cA2[it]);
    }
#pragma unroll
    for (int it = 0; it < 4; ++it)
        wreg[it] = *(const float4*)(W + (size_t)rW[it] * N + nWG + nW[it]);

    v8f acc[2][4];
#pragma unroll
    for (int i = 0; i < 2; ++i)
#pragma unroll
        for (int j = 0; j < 4; ++j) acc[i][j] = v8f{};

    int buf = 0;
    for (int k0 = 0; k0 < K; k0 += 32) {
        // ---- commit staged registers to LDS[buf] ----
        if (A_F32) {
#pragma unroll
            for (int it = 0; it < 4; ++it) {
                *(unsigned int*)&As[buf][rA4[it]][cA4[it]] =
                    pack_bf16_pair(aregF[it].x, aregF[it].y);
                *(unsigned int*)&As[buf][rA4[it]][cA4[it] + 2] =
                    pack_bf16_pair(aregF[it].z, aregF[it].w);
            }
        } else {
#pragma unroll
            for (int it = 0; it < 8; ++it)
                *(unsigned int*)&As[buf][rA2[it]][cA2[it]] = aregB[it];
        }
#pragma unroll
        for (int it = 0; it < 4; ++it) {
            Bs[buf][nW[it] + 0][rW[it]] = (__bf16)wreg[it].x;
            Bs[buf][nW[it] + 1][rW[it]] = (__bf16)wreg[it].y;
            Bs[buf][nW[it] + 2][rW[it]] = (__bf16)wreg[it].z;
            Bs[buf][nW[it] + 3][rW[it]] = (__bf16)wreg[it].w;
        }
        __syncthreads();   // LDS[buf] ready; also fences iteration k-1's reads

        // ---- prefetch next tile into registers (overlaps with WMMAs) ----
        const int k1 = k0 + 32;
        if (k1 < K) {
            if (A_F32) {
#pragma unroll
                for (int it = 0; it < 4; ++it)
                    aregF[it] = *(const float4*)(Af + (size_t)(mWG + rA4[it]) * K + k1 + cA4[it]);
            } else {
#pragma unroll
                for (int it = 0; it < 8; ++it)
                    aregB[it] = *(const unsigned int*)(Abf + (size_t)(mWG + rA2[it]) * K + k1 + cA2[it]);
            }
#pragma unroll
            for (int it = 0; it < 4; ++it)
                wreg[it] = *(const float4*)(W + (size_t)(k1 + rW[it]) * N + nWG + nW[it]);
        }

        // ---- compute from LDS[buf] ----
        v16bf a[2], b[4];
#pragma unroll
        for (int i = 0; i < 2; ++i)
            a[i] = gather_a(&As[buf][wm * 32 + i * 16 + l15][0], half);
#pragma unroll
        for (int j = 0; j < 4; ++j)
            b[j] = gather_b(&Bs[buf][wn * 64 + j * 16 + l15][half * 16]);
#pragma unroll
        for (int i = 0; i < 2; ++i)
#pragma unroll
            for (int j = 0; j < 4; ++j)
                acc[i][j] = wmma_bf16(a[i], b[j], acc[i][j]);

        buf ^= 1;
    }

    // ---- epilogue: bias, scale, scatter ----
#pragma unroll
    for (int i = 0; i < 2; ++i) {
#pragma unroll
        for (int j = 0; j < 4; ++j) {
            const int n    = nWG + wn * 64 + j * 16 + l15;
            const float bv = bias[n];
#pragma unroll
            for (int r = 0; r < 8; ++r) {
                const int m   = mWG + wm * 32 + i * 16 + half * 8 + r;
                const float v = (acc[i][j][r] + bv) * scale;
                if (HEAD_MAJOR) {
                    const int bb = m / S, s = m % S;
                    const int h = n >> 6, d = n & 63;
                    out_bf[((size_t)(bb * 16 + h) * S + s) * 64 + d] = (__bf16)v;
                } else {
                    out_f[(size_t)m * N + n] = v;
                }
            }
        }
    }
}

// ---------------------------------------------------------------------------
// Flash attention. Grid: (S/128, B*H). 8 waves; wave w owns q-rows [16w,16w+16).
// K-tile = 64 keys per iteration. dh = 64. K/V/mask register-prefetched.
// ---------------------------------------------------------------------------
__global__ __launch_bounds__(256) void attn_kernel(
    const __bf16* __restrict__ Qbf, const __bf16* __restrict__ Kbf,
    const __bf16* __restrict__ Vbf, const int* __restrict__ mask,
    __bf16* __restrict__ ctx, int Slen)
{
    __shared__ __bf16 Qs [128][66];  // [q][dh]
    __shared__ __bf16 KPs[128][66];  // rows 0..63: K tile [key][dh]; reused as P [q][key]
    __shared__ __bf16 Vts[64][66];   // V transposed: [dh][key]
    __shared__ float  mAdd[64];      // additive key mask (0 or -1e20)

    const int tid  = threadIdx.x;
    const int lane = tid & 31;
    const int wid  = tid >> 5;
    const int half = lane >> 4;
    const int l15  = lane & 15;

    const int bh = blockIdx.y;         // b*16 + h
    const int b  = bh >> 4;
    const int h  = bh & 15;
    const int q0 = blockIdx.x * 128;

    const __bf16* Qp = Qbf + (size_t)bh * Slen * 64;
    const __bf16* Kp = Kbf + (size_t)bh * Slen * 64;
    const __bf16* Vp = Vbf + (size_t)bh * Slen * 64;

    // per-thread staging coordinates for 64x64 tiles (pair granularity)
    int rT[8], cT[8];
#pragma unroll
    for (int it = 0; it < 8; ++it) {
        int idx = tid + it * 256;
        rT[it] = idx >> 5;          // key row 0..63
        cT[it] = (idx & 31) * 2;    // dh pair
    }

    // ---- stage Q tile (128x64), persistent across key loop ----
#pragma unroll
    for (int it = 0; it < 16; ++it) {
        int idx = tid + it * 256;        // 0..4095 bf16 pairs
        int r   = idx >> 5;
        int c   = (idx & 31) * 2;
        *(unsigned int*)&Qs[r][c] =
            *(const unsigned int*)(Qp + (size_t)(q0 + r) * 64 + c);
    }

    v8f   oacc[4];
    float m_i[8], l_i[8];
#pragma unroll
    for (int j = 0; j < 4; ++j) oacc[j] = v8f{};
#pragma unroll
    for (int r = 0; r < 8; ++r) { m_i[r] = -3e38f; l_i[r] = 0.0f; }

    // ---- prologue: stage K/V/mask tile 0 into registers ----
    unsigned int kreg[8], vreg[8];
    float mreg = 0.0f;
#pragma unroll
    for (int it = 0; it < 8; ++it) {
        kreg[it] = *(const unsigned int*)(Kp + (size_t)rT[it] * 64 + cT[it]);
        vreg[it] = *(const unsigned int*)(Vp + (size_t)rT[it] * 64 + cT[it]);
    }
    if (tid < 64) mreg = (mask[b * Slen + tid] == 0) ? -1e20f : 0.0f;

    const int nT = Slen / 64;
    for (int kt = 0; kt < nT; ++kt) {
        // ---- commit staged K/V/mask to LDS ----
#pragma unroll
        for (int it = 0; it < 8; ++it)
            *(unsigned int*)&KPs[rT[it]][cT[it]] = kreg[it];
#pragma unroll
        for (int it = 0; it < 8; ++it) {
            Vts[cT[it] + 0][rT[it]] =
                __builtin_bit_cast(__bf16, (unsigned short)(vreg[it] & 0xffffu));
            Vts[cT[it] + 1][rT[it]] =
                __builtin_bit_cast(__bf16, (unsigned short)(vreg[it] >> 16));
        }
        if (tid < 64) mAdd[tid] = mreg;
        __syncthreads();                               // (1) tiles visible

        // ---- S = Q . K^T  (wave strip: 16 q-rows x 64 keys) ----
        v8f sacc[4];
#pragma unroll
        for (int j = 0; j < 4; ++j) sacc[j] = v8f{};
#pragma unroll
        for (int ks = 0; ks < 2; ++ks) {
            v16bf aq = gather_a(&Qs[wid * 16 + l15][ks * 32], half);
#pragma unroll
            for (int j = 0; j < 4; ++j) {
                v16bf bk = gather_b(&KPs[j * 16 + l15][ks * 32 + half * 16]);
                sacc[j] = wmma_bf16(aq, bk, sacc[j]);
            }
        }

        // ---- prefetch next K/V/mask tile (in flight during softmax + PV) ----
        if (kt + 1 < nT) {
            const int kb1 = (kt + 1) * 64;
#pragma unroll
            for (int it = 0; it < 8; ++it) {
                kreg[it] = *(const unsigned int*)(Kp + (size_t)(kb1 + rT[it]) * 64 + cT[it]);
                vreg[it] = *(const unsigned int*)(Vp + (size_t)(kb1 + rT[it]) * 64 + cT[it]);
            }
            if (tid < 64) mreg = (mask[b * Slen + kb1 + tid] == 0) ? -1e20f : 0.0f;
        }
        __syncthreads();                               // (2) K-tile reads done

        // ---- online softmax; write P into KPs (wave-private 16-row block) ----
#pragma unroll
        for (int r = 0; r < 8; ++r) {
            float mx = -3e38f;
#pragma unroll
            for (int j = 0; j < 4; ++j) {
                float s = sacc[j][r] + mAdd[j * 16 + l15];
                sacc[j][r] = s;
                mx = fmaxf(mx, s);
            }
#pragma unroll
            for (int xm = 1; xm < 16; xm <<= 1)
                mx = fmaxf(mx, __shfl_xor(mx, xm, 32));   // within 16-lane half
            const float mnew  = fmaxf(m_i[r], mx);
            const float alpha = __expf(m_i[r] - mnew);
            m_i[r] = mnew;
            l_i[r] *= alpha;

            float rs = 0.0f;
            const int prow = wid * 16 + half * 8 + r;
#pragma unroll
            for (int j = 0; j < 4; ++j) {
                float p = __expf(sacc[j][r] - mnew);
                rs += p;
                KPs[prow][j * 16 + l15] = (__bf16)p;
            }
#pragma unroll
            for (int xm = 1; xm < 16; xm <<= 1)
                rs += __shfl_xor(rs, xm, 32);
            l_i[r] += rs;
#pragma unroll
            for (int j = 0; j < 4; ++j) oacc[j][r] *= alpha;
        }
        // P rows are produced and consumed by the SAME wave; LDS ops are
        // in-order per wave, so a compiler fence (no hw barrier) suffices.
        asm volatile("" ::: "memory");

        // ---- O += P . V ----
#pragma unroll
        for (int ks = 0; ks < 2; ++ks) {
            v16bf ap = gather_a(&KPs[wid * 16 + l15][ks * 32], half);
#pragma unroll
            for (int j = 0; j < 4; ++j) {
                v16bf bv = gather_b(&Vts[j * 16 + l15][ks * 32 + half * 16]);
                oacc[j] = wmma_bf16(ap, bv, oacc[j]);
            }
        }
        __syncthreads();                               // (3) P/Vts reads done
    }

    // ---- normalize rows, write ctx [token][H*dh] as bf16 ----
#pragma unroll
    for (int r = 0; r < 8; ++r) {
        const int   m   = wid * 16 + half * 8 + r;
        const float inv = (l_i[r] > 0.0f) ? (1.0f / l_i[r]) : 0.0f;
        const size_t base = ((size_t)(b * Slen + q0 + m)) * 1024 + h * 64;
#pragma unroll
        for (int j = 0; j < 4; ++j)
            ctx[base + j * 16 + l15] = (__bf16)(oacc[j][r] * inv);
    }
}

// ---------------------------------------------------------------------------
// Launch. Inputs: query,key,value,mask,Wq,bq,Wk,bk,Wv,bv,Wo,bo
// ---------------------------------------------------------------------------
extern "C" void kernel_launch(void* const* d_in, const int* in_sizes, int n_in,
                              void* d_out, int out_size, void* d_ws, size_t ws_size,
                              hipStream_t stream) {
    (void)in_sizes; (void)n_in; (void)out_size; (void)ws_size;

    const float* query = (const float*)d_in[0];
    const float* keyin = (const float*)d_in[1];
    const float* value = (const float*)d_in[2];
    const int*   mask  = (const int*)d_in[3];
    const float* Wq = (const float*)d_in[4];
    const float* bq = (const float*)d_in[5];
    const float* Wk = (const float*)d_in[6];
    const float* bk = (const float*)d_in[7];
    const float* Wv = (const float*)d_in[8];
    const float* bv = (const float*)d_in[9];
    const float* Wo = (const float*)d_in[10];
    const float* bo = (const float*)d_in[11];

    const int B = 4, S = 2048, D = 1024;
    const int M = B * S;                          // 8192 tokens
    const size_t planeBytes = (size_t)M * D * 2;  // 16 MB per bf16 plane

    char* ws = (char*)d_ws;
    __bf16* Qbf = (__bf16*)(ws + 0 * planeBytes);
    __bf16* Kbf = (__bf16*)(ws + 1 * planeBytes);
    __bf16* Vbf = (__bf16*)(ws + 2 * planeBytes);
    __bf16* Ctx = (__bf16*)(ws + 3 * planeBytes);

    dim3 blk(256);
    dim3 gProj(D / 128, M / 128);                 // (8, 64)

    // scale = 1/sqrt(dh) = 1/8 folded into the Q projection
    gemm_bf16_kernel<true, true><<<gProj, blk, 0, stream>>>(
        query, nullptr, Wq, bq, Qbf, nullptr, M, D, D, 0.125f, S);
    gemm_bf16_kernel<true, true><<<gProj, blk, 0, stream>>>(
        keyin, nullptr, Wk, bk, Kbf, nullptr, M, D, D, 1.0f, S);
    gemm_bf16_kernel<true, true><<<gProj, blk, 0, stream>>>(
        value, nullptr, Wv, bv, Vbf, nullptr, M, D, D, 1.0f, S);

    dim3 gAttn(S / 128, B * 16);                  // (16, 64)
    attn_kernel<<<gAttn, blk, 0, stream>>>(Qbf, Kbf, Vbf, mask, Ctx, S);

    gemm_bf16_kernel<false, false><<<gProj, blk, 0, stream>>>(
        nullptr, Ctx, Wo, bo, nullptr, (float*)d_out, M, D, D, 1.0f, S);
}
